// LocalAttention_63101659513214
// MI455X (gfx1250) — matrix operand order repaired and checked by reading, not compiled, not run on hardware
//
#include <hip/hip_runtime.h>

typedef _Float16 h16;
typedef _Float16 v4h  __attribute__((ext_vector_type(4)));
typedef _Float16 v8h  __attribute__((ext_vector_type(8)));
typedef _Float16 v16h __attribute__((ext_vector_type(16)));
typedef float    v8f  __attribute__((ext_vector_type(8)));

namespace {
constexpr int kB = 4, kH = 16, kT = 4096, kD = 64;
constexpr int kWS = 128;                   // window size
constexpr int kNW = kT / kWS;              // 32 windows
constexpr int kKW = 2 * kWS;               // 256 keys visible per window (look-back 1)
constexpr int kKStride = kD + 8;           // 72 halves per Ksh row (bank-conflict pad)
constexpr int kVStride = kKW + 8;          // 264 halves per Vt row
constexpr int kPStride = kKW + 8;          // 264 halves per P row
constexpr size_t kSmemHalves =
    (size_t)kKW * kKStride + (size_t)kD * kVStride + (size_t)8 * 16 * kPStride;
constexpr float kFltMax = 3.402823466e38f; // mask value = -kFltMax (matches reference)
}

union V16u { v16h v; v8h h[2]; h16 e[16]; };

__global__ __launch_bounds__(256)
void local_attn_wmma(const float* __restrict__ q, const float* __restrict__ k,
                     const float* __restrict__ v, float* __restrict__ out,
                     float* __restrict__ prev)
{
  extern __shared__ h16 smem[];
  h16* Ksh = smem;                          // [256][72]  key rows (f16)
  h16* Vt  = Ksh + kKW * kKStride;          // [64][264]  V transposed (f16)
  h16* Psh = Vt + kD * kVStride;            // [8 waves][16][264] exp scores (f16)

  const int bh   = blockIdx.x / kNW;
  const int w    = blockIdx.x % kNW;
  const int tid  = threadIdx.x;
  const int lane = tid & 31;
  const int wave = tid >> 5;
  const int ln   = lane & 15;               // N / M-sub index
  const int hh   = lane >> 4;               // lane half

  const size_t baseQ = (size_t)bh * kT * kD;

  // ---- K staging: one key row per thread, packed v8h -> ds_store_b128 x8 ----
  {
    const int j  = tid;                     // key slot 0..255
    const int tk = (w - 1) * kWS + j;       // absolute key time
    const bool pad = (tk < 0);              // w==0 look-back window -> zeros
    const float4* ks = (const float4*)(k + baseQ + (size_t)(pad ? 0 : tk) * kD);
    h16* krow = Ksh + j * kKStride;
#pragma unroll
    for (int c = 0; c < 8; ++c) {
      float4 a = ks[2 * c], b = ks[2 * c + 1];
      if (pad) { a.x = a.y = a.z = a.w = 0.f; b.x = b.y = b.z = b.w = 0.f; }
      v8h pk;
      pk[0] = (h16)a.x; pk[1] = (h16)a.y; pk[2] = (h16)a.z; pk[3] = (h16)a.w;
      pk[4] = (h16)b.x; pk[5] = (h16)b.y; pk[6] = (h16)b.z; pk[7] = (h16)b.w;
      *(v8h*)(krow + 8 * c) = pk;
    }
  }

  // ---- V staging (transposed): 4 rows x 16 cols per thread, b64 packed stores ----
  {
    const int db = tid & 3;                 // d cols db*16 .. db*16+15
    const int jb = tid >> 2;                // j rows 4*jb .. 4*jb+3
    h16 hv[4][16];
#pragma unroll
    for (int jj = 0; jj < 4; ++jj) {
      const int j  = 4 * jb + jj;
      const int tk = (w - 1) * kWS + j;
      const bool pad = (tk < 0);
      const float* vr = v + baseQ + (size_t)(pad ? 0 : tk) * kD + db * 16;
#pragma unroll
      for (int x = 0; x < 4; ++x) {
        float4 a = *(const float4*)(vr + 4 * x);
        if (pad) { a.x = a.y = a.z = a.w = 0.f; }
        hv[jj][4 * x + 0] = (h16)a.x; hv[jj][4 * x + 1] = (h16)a.y;
        hv[jj][4 * x + 2] = (h16)a.z; hv[jj][4 * x + 3] = (h16)a.w;
      }
    }
#pragma unroll
    for (int d = 0; d < 16; ++d) {
      v4h pk;
      pk[0] = hv[0][d]; pk[1] = hv[1][d]; pk[2] = hv[2][d]; pk[3] = hv[3][d];
      *(v4h*)(Vt + (db * 16 + d) * kVStride + 4 * jb) = pk;
    }
  }

  // ---- per-wave Q fragments (A layout 16x32 f16: K = (e/8)*16 + hh*8 + e%8) ----
  const int qi = wave * 16 + ln;            // query row within window (A: M = lane%16)
  const float* qrow = q + baseQ + (size_t)(w * kWS + qi) * kD;
  V16u qa[2];
#pragma unroll
  for (int kv = 0; kv < 2; ++kv) {
    const float* p0 = qrow + kv * 32 + hh * 8;
#pragma unroll
    for (int e = 0; e < 8; ++e) {
      qa[kv].e[e]     = (h16)p0[e];
      qa[kv].e[e + 8] = (h16)p0[16 + e];
    }
  }

  __syncthreads();

  // ---- S = Q K^T, software-pipelined: tile nt+1's WMMAs issue before tile nt's
  //      post-processing consumes its accumulator (absorbs WMMA->VALU hazard). ----
  float rsum[8];
#pragma unroll
  for (int r = 0; r < 8; ++r) rsum[r] = 0.f;

  // Row-base pointers: per-r accesses become constant immediate offsets.
  float* prevb0 = prev + ((size_t)bh * kNW + w) * kWS * kKW
                       + (size_t)(wave * 16 + hh * 8) * kKW + ln;
  h16* prow0 = Psh + (wave * 16 + hh * 8) * kPStride + ln;

  const h16* kfrag0 = Ksh + ln * kKStride + hh * 8;  // + nt*16*kKStride + kv*32

#define S_TILE(dst, nt_)                                                        \
  do {                                                                          \
    v8f c_ = {};                                                                \
    _Pragma("unroll")                                                           \
    for (int kv = 0; kv < 2; ++kv) {                                            \
      const h16* kp = kfrag0 + (nt_) * 16 * kKStride + kv * 32;                 \
      V16u bf;                                                                  \
      bf.h[0] = *(const v8h*)kp;                                                \
      bf.h[1] = *(const v8h*)(kp + 16);                                         \
      c_ = __builtin_amdgcn_wmma_f32_16x16x32_f16(false, qa[kv].v, false, bf.v, \
                                                  (short)0, c_, false, false);  \
    }                                                                           \
    (dst) = c_;                                                                 \
  } while (0)

  v8f acc;
  S_TILE(acc, 0);
#pragma unroll
  for (int nt = 0; nt < 16; ++nt) {
    v8f nxt = {};
    if (nt < 15) S_TILE(nxt, nt + 1);       // independent work over hazard window

    const int jc = nt * 16 + ln;            // key slot 0..255
    float sv[8];
#pragma unroll
    for (int r = 0; r < 8; ++r) {
      const int il = wave * 16 + hh * 8 + r;      // query row (C/D: M = r + 8*hh)
      const float s = acc[r] * 0.125f;            // D^-0.5 = 1/8
      const bool masked = (nt < 8) ? (w == 0)     // pad mask (look-back of window 0)
                                   : (il < jc - kWS);  // causal, current window
      sv[r] = masked ? -kFltMax : s;
    }
    float* pb = prevb0 + nt * 16;
#pragma unroll
    for (int r = 0; r < 8; ++r) pb[r * kKW] = sv[r];   // imm offsets r*1024B

    float pe[8];
#pragma unroll
    for (int r = 0; r < 8; ++r) pe[r] = __expf(sv[r]); // masked -> 0; chains interleave
#pragma unroll
    for (int r = 0; r < 8; ++r) rsum[r] += pe[r];
    h16 ph[8];
#pragma unroll
    for (int r = 0; r < 8; ++r) ph[r] = (h16)pe[r];
    h16* pp = prow0 + nt * 16;
#pragma unroll
    for (int r = 0; r < 8; ++r) pp[r * kPStride] = ph[r];  // imm offsets r*528B

    acc = nxt;
  }
#undef S_TILE

  // ---- row sums: reduce across the 16 lanes holding each row; take reciprocal ----
#pragma unroll
  for (int r = 0; r < 8; ++r) {
    float sum = rsum[r];
    sum += __shfl_xor(sum, 1, 32);
    sum += __shfl_xor(sum, 2, 32);
    sum += __shfl_xor(sum, 4, 32);
    sum += __shfl_xor(sum, 8, 32);
    rsum[r] = 1.0f / sum;                   // > 0, diagonal always present
  }

  // ---- O = P V. P is wave-private in LDS (same-wave dscnt ordering suffices). ----
  V16u af[8];                               // hoist all A fragments (P rows) once
#pragma unroll
  for (int kv = 0; kv < 8; ++kv) {
    const h16* pp = Psh + (wave * 16 + ln) * kPStride + kv * 32 + hh * 8;
    af[kv].h[0] = *(const v8h*)pp;          // j = kv*32 + hh*8 + e
    af[kv].h[1] = *(const v8h*)(pp + 16);   // j = kv*32 + 16 + hh*8 + e
  }

  const h16* vfrag0 = Vt + ln * kVStride + hh * 8;   // + nt*16*kVStride + kv*32

#define PV_TILE(dst, nt_)                                                        \
  do {                                                                           \
    v8f c_ = {};                                                                 \
    _Pragma("unroll")                                                            \
    for (int kv = 0; kv < 8; ++kv) {                                             \
      const h16* vp = vfrag0 + (nt_) * 16 * kVStride + kv * 32;                  \
      V16u bf;                                                                   \
      bf.h[0] = *(const v8h*)vp;                                                 \
      bf.h[1] = *(const v8h*)(vp + 16);                                          \
      c_ = __builtin_amdgcn_wmma_f32_16x16x32_f16(false, af[kv].v, false, bf.v,  \
                                                  (short)0, c_, false, false);   \
    }                                                                            \
    (dst) = c_;                                                                  \
  } while (0)

  float* ob = out + baseQ + (size_t)(w * kWS + wave * 16 + hh * 8) * kD + ln;
  v8f oacc;
  PV_TILE(oacc, 0);
#pragma unroll
  for (int nt = 0; nt < 4; ++nt) {
    v8f nxt = {};
    if (nt < 3) PV_TILE(nxt, nt + 1);       // overlap next tile with normalization
#pragma unroll
    for (int r = 0; r < 8; ++r)
      ob[(size_t)r * kD + nt * 16] = oacc[r] * rsum[r];  // imm offsets r*256B
    oacc = nxt;
  }
#undef PV_TILE
}

extern "C" void kernel_launch(void* const* d_in, const int* in_sizes, int n_in,
                              void* d_out, int out_size, void* d_ws, size_t ws_size,
                              hipStream_t stream)
{
  (void)in_sizes; (void)n_in; (void)out_size; (void)d_ws; (void)ws_size;
  const float* q = (const float*)d_in[0];
  const float* k = (const float*)d_in[1];
  const float* v = (const float*)d_in[2];
  float* out  = (float*)d_out;
  float* prev = out + (size_t)kB * kH * kT * kD;   // tuple order: (out, prev)

  const dim3 grid(kB * kH * kNW);                  // 2048 tiles, one per workgroup
  const size_t smem = kSmemHalves * sizeof(h16);   // ~135 KB dynamic LDS
  local_attn_wmma<<<grid, dim3(256), smem, stream>>>(q, k, v, out, prev);
}
